// HANLayer_41420664602834
// MI455X (gfx1250) — matrix operand order repaired and checked by reading, not compiled, not run on hardware
//
#include <hip/hip_runtime.h>
#include <hip/hip_bf16.h>
#include <math.h>

#define NN     100000
#define IN_F   256
#define HH     4
#define DD     32
#define HDIM   128
#define PPATH  3
#define EDGES  600000
#define SAHID  128

typedef __attribute__((ext_vector_type(16))) __bf16   v16bf;
typedef __attribute__((ext_vector_type(8)))  float    v8f;
typedef __attribute__((ext_vector_type(8)))  unsigned v8u;

// ---- float <-> bf16 (RNE) helpers for the hi/lo split-precision WMMA path ----
static __device__ __forceinline__ __bf16 f2bf(float f) {
    unsigned u = __float_as_uint(f);
    unsigned r = u + 0x7FFFu + ((u >> 16) & 1u);
    unsigned short s = (unsigned short)(r >> 16);
    return __builtin_bit_cast(__bf16, s);
}
static __device__ __forceinline__ float bf2f(__bf16 b) {
    unsigned short s = __builtin_bit_cast(unsigned short, b);
    return __uint_as_float(((unsigned)s) << 16);
}
static __device__ __forceinline__ unsigned pack2bf(__bf16 a, __bf16 b) {
    return (unsigned)__builtin_bit_cast(unsigned short, a) |
           ((unsigned)__builtin_bit_cast(unsigned short, b) << 16);
}

// ---- order-preserving float<->uint encoding for atomicMax-based segment max ----
static __device__ __forceinline__ unsigned fenc(float f) {
    unsigned u = __float_as_uint(f);
    return (u & 0x80000000u) ? ~u : (u | 0x80000000u);
}
static __device__ __forceinline__ float fdec(unsigned k) {
    unsigned u = (k & 0x80000000u) ? (k & 0x7FFFFFFFu) : ~k;
    return __uint_as_float(u);
}

// ---------------- fill ----------------
__global__ void fill_u32(unsigned* __restrict__ p, long long n, unsigned v) {
    long long i = (long long)blockIdx.x * blockDim.x + threadIdx.x;
    long long stride = (long long)gridDim.x * blockDim.x;
    for (; i < n; i += stride) p[i] = v;
}

// ---------------- pack W into fragment-ordered bf16 hi/lo panels ----------------
// dword t = [p][ct(8)][ks(8)][lane(32)][j(8)] ; value = packed (bf16(W[k,n]), bf16(W[k+1,n]))
// with k = ks*32 + (lane>>4)*16 + 2j , n = ct*16 + (lane&15)  (contiguous k order, same as the A fragments)
#define WPACK_N (PPATH * 8 * 8 * 32 * 8)
__global__ void pack_W(const float* __restrict__ W,
                       unsigned* __restrict__ Whi, unsigned* __restrict__ Wlo) {
    int t = blockIdx.x * blockDim.x + threadIdx.x;
    if (t >= WPACK_N) return;
    int j    = t & 7;
    int lane = (t >> 3) & 31;
    int ks   = (t >> 8) & 7;
    int ct   = (t >> 11) & 7;
    int p    = t >> 14;
    int k = ks * 32 + ((lane >> 4) * 16) + 2 * j;
    int n = ct * 16 + (lane & 15);
    float b0 = W[((size_t)p * IN_F + k) * HDIM + n];
    float b1 = W[((size_t)p * IN_F + k + 1) * HDIM + n];
    __bf16 h0 = f2bf(b0), h1 = f2bf(b1);
    Whi[t] = pack2bf(h0, h1);
    Wlo[t] = pack2bf(f2bf(b0 - bf2f(h0)), f2bf(b1 - bf2f(h1)));
}

// same idea for sa_w1: t = [ct(8)][kk(4)][lane(32)][j(8)]
#define S1PACK_N (8 * 4 * 32 * 8)
__global__ void pack_S1(const float* __restrict__ w1,
                        unsigned* __restrict__ S1hi, unsigned* __restrict__ S1lo) {
    int t = blockIdx.x * blockDim.x + threadIdx.x;
    if (t >= S1PACK_N) return;
    int j    = t & 7;
    int lane = (t >> 3) & 31;
    int kk   = (t >> 8) & 3;
    int ct   = t >> 10;
    int k = kk * 32 + ((lane >> 4) * 16) + 2 * j;
    int n = ct * 16 + (lane & 15);
    float b0 = w1[(size_t)k * SAHID + n];
    float b1 = w1[(size_t)(k + 1) * SAHID + n];
    __bf16 h0 = f2bf(b0), h1 = f2bf(b1);
    S1hi[t] = pack2bf(h0, h1);
    S1lo[t] = pack2bf(f2bf(b0 - bf2f(h0)), f2bf(b1 - bf2f(h1)));
}

// ---------------- feat = h @ W[p]  (bf16 hi/lo split WMMA, f32 accum) ----------------
// block = 256 threads = 8 waves; block computes 16 rows x 128 cols; wave w owns cols [16w,16w+16)
__global__ __launch_bounds__(256) void gemm_feat(const float* __restrict__ h,
                                                 const unsigned* __restrict__ Whi,
                                                 const unsigned* __restrict__ Wlo,
                                                 float* __restrict__ feat) {
    __shared__ __bf16 a_hi_lds[16 * 264];   // padded row stride to dodge bank conflicts
    __shared__ __bf16 a_lo_lds[16 * 264];
    const int p   = blockIdx.y;
    const int m0  = blockIdx.x * 16;
    const int tid = threadIdx.x;

    // stage A tile (16 x 256 fp32) into LDS as bf16 hi/lo, coalesced (once per element)
    for (int i = 0; i < 16; ++i) {
        int idx = tid + i * 256;            // 0..4095
        int r = idx >> 8;
        int k = idx & 255;
        float v = h[(size_t)(m0 + r) * IN_F + k];
        __bf16 hi = f2bf(v);
        a_hi_lds[r * 264 + k] = hi;
        a_lo_lds[r * 264 + k] = f2bf(v - bf2f(hi));
    }
    __syncthreads();

    const int wave = tid >> 5;              // = col tile ct
    const int lane = tid & 31;
    const int n0   = wave * 16;
    const int mrow = lane & 15;             // A: M = lane%16
    const int kbA  = (lane >> 4) * 8;       // A: lane half selects K-subgroup
    const int ncol = lane & 15;             // C/D: N = lane%16

    v8f c = {};
    // fragment-ordered B panels: [(p*8+ct)*8 + ks] blocks of 32 lanes x 8 dwords
    const unsigned* WhiP = Whi + ((((size_t)p * 8 + wave) * 8) * 32 + lane) * 8;
    const unsigned* WloP = Wlo + ((((size_t)p * 8 + wave) * 8) * 32 + lane) * 8;

#pragma unroll
    for (int ks = 0; ks < 8; ++ks) {
        if (ks + 1 < 8)
            __builtin_prefetch(WhiP + (ks + 1) * 256, 0, 0);
        v16bf ah, al;
#pragma unroll
        for (int j = 0; j < 8; ++j) {
            // A 16x32 bf16 layout: vgpr j holds K = 2*kbA + 2*j (+1)
            int kloc = ks * 32 + 2 * kbA + 2 * j;
            ah[2 * j]     = a_hi_lds[mrow * 264 + kloc];
            ah[2 * j + 1] = a_hi_lds[mrow * 264 + kloc + 1];
            al[2 * j]     = a_lo_lds[mrow * 264 + kloc];
            al[2 * j + 1] = a_lo_lds[mrow * 264 + kloc + 1];
        }
        v16bf bh = __builtin_bit_cast(v16bf, *(const v8u*)(WhiP + ks * 256));
        v16bf bl = __builtin_bit_cast(v16bf, *(const v8u*)(WloP + ks * 256));
        c = __builtin_amdgcn_wmma_f32_16x16x32_bf16(false, ah, false, bh, (short)0, c, false, false);
        c = __builtin_amdgcn_wmma_f32_16x16x32_bf16(false, ah, false, bl, (short)0, c, false, false);
        c = __builtin_amdgcn_wmma_f32_16x16x32_bf16(false, al, false, bh, (short)0, c, false, false);
    }

    float* Fp = feat + (size_t)p * NN * HDIM;
#pragma unroll
    for (int r = 0; r < 8; ++r) {
        int m = m0 + r + ((lane >> 4) * 8);   // D layout: vgpr r -> M=r (lanes<16) / r+8
        Fp[(size_t)m * HDIM + n0 + ncol] = c[r];
    }
}

// ---------------- per-node attention logits el/er ----------------
__global__ void el_er_kernel(const float* __restrict__ feat,
                             const float* __restrict__ al, const float* __restrict__ ar,
                             float* __restrict__ el, float* __restrict__ er) {
    int t = blockIdx.x * blockDim.x + threadIdx.x;
    if (t >= PPATH * NN * HH) return;
    int hh = t % HH;
    int n  = (t / HH) % NN;
    int p  = t / (NN * HH);
    const float* f   = feat + ((size_t)p * NN + n) * HDIM + hh * DD;
    const float* alp = al + (p * HH + hh) * DD;
    const float* arp = ar + (p * HH + hh) * DD;
    float sl = 0.f, sr = 0.f;
    for (int d = 0; d < DD; ++d) { float v = f[d]; sl += v * alp[d]; sr += v * arp[d]; }
    el[t] = sl; er[t] = sr;                 // layout [p][n][h]
}

// ---------------- edge leaky-relu + segment max (encoded uint atomicMax) ----------------
__global__ void edge_max_kernel(const int* __restrict__ src, const int* __restrict__ dst,
                                const float* __restrict__ el, const float* __restrict__ er,
                                unsigned* __restrict__ mkey) {
    int t = blockIdx.x * blockDim.x + threadIdx.x;
    if (t >= PPATH * EDGES) return;
    int p = t / EDGES;
    int s = src[t], d = dst[t];
    for (int hh = 0; hh < HH; ++hh) {
        float v = el[((size_t)p * NN + s) * HH + hh] + er[((size_t)p * NN + d) * HH + hh];
        v = (v >= 0.f) ? v : 0.2f * v;
        atomicMax(&mkey[((size_t)p * NN + d) * HH + hh], fenc(v));
    }
}

// ---------------- ex = exp(e - max); denom = segment_sum(ex) ----------------
__global__ void edge_exp_kernel(const int* __restrict__ src, const int* __restrict__ dst,
                                const float* __restrict__ el, const float* __restrict__ er,
                                const unsigned* __restrict__ mkey,
                                float* __restrict__ exbuf, float* __restrict__ denom) {
    int t = blockIdx.x * blockDim.x + threadIdx.x;
    if (t >= PPATH * EDGES) return;
    int p = t / EDGES;
    int s = src[t], d = dst[t];
    for (int hh = 0; hh < HH; ++hh) {
        float v = el[((size_t)p * NN + s) * HH + hh] + er[((size_t)p * NN + d) * HH + hh];
        v = (v >= 0.f) ? v : 0.2f * v;
        float m = fdec(mkey[((size_t)p * NN + d) * HH + hh]);
        float ex = expf(v - m);
        exbuf[(size_t)t * HH + hh] = ex;
        atomicAdd(&denom[((size_t)p * NN + d) * HH + hh], ex);
    }
}

// ---------------- weighted scatter: rst[dst] += feat[src] * alpha ----------------
__global__ void message_kernel(const int* __restrict__ src, const int* __restrict__ dst,
                               const float* __restrict__ feat, const float* __restrict__ exbuf,
                               const float* __restrict__ denom, float* __restrict__ rst) {
    long long t = (long long)blockIdx.x * blockDim.x + threadIdx.x;
    if (t >= (long long)PPATH * EDGES * HDIM) return;
    int dim = (int)(t % HDIM);
    long long pe = t / HDIM;
    int p = (int)(pe / EDGES);
    int hh = dim >> 5;
    int s = src[pe], d = dst[pe];
    float ex = exbuf[pe * HH + hh];
    float dn = denom[((size_t)p * NN + d) * HH + hh];
    float alpha = ex / fmaxf(dn, 1e-9f);
    atomicAdd(&rst[((size_t)p * NN + d) * HDIM + dim],
              feat[((size_t)p * NN + s) * HDIM + dim] * alpha);
}

// ---------------- z = elu(rst + bias) in place ----------------
__global__ void elu_kernel(float* __restrict__ z, const float* __restrict__ bias) {
    long long t = (long long)blockIdx.x * blockDim.x + threadIdx.x;
    if (t >= (long long)PPATH * NN * HDIM) return;
    int dim = (int)(t % HDIM);
    int p = (int)(t / ((long long)NN * HDIM));
    float v = z[t] + bias[p * HDIM + dim];
    z[t] = (v > 0.f) ? v : expm1f(v);
}

// ---------------- semantic attention: wsum[p] += sum_rows (tanh(z@w1+b1)@w2) ----------------
// one wave per 16-row tile of z[p]; 8 col-tile accumulators; kk-outer; WMMA bf16 hi/lo
__global__ __launch_bounds__(256) void sa_kernel(const float* __restrict__ z,
                                                 const unsigned* __restrict__ S1hi,
                                                 const unsigned* __restrict__ S1lo,
                                                 const float* __restrict__ b1,
                                                 const float* __restrict__ w2,
                                                 float* __restrict__ wsum) {
    int wid = blockIdx.x * 8 + (threadIdx.x >> 5);
    if (wid >= PPATH * (NN / 16)) return;       // whole-wave uniform exit (EXEC stays all-1)
    int p  = wid / (NN / 16);
    int m0 = (wid % (NN / 16)) * 16;
    int lane = threadIdx.x & 31;
    const int mrow = lane & 15, kbA = (lane >> 4) * 8;
    const int ncol = lane & 15;

    v8f acc[8] = {};
    const float* zp = z + (size_t)p * NN * HDIM;

    for (int kk = 0; kk < 4; ++kk) {            // K = 128 in 4 steps of 32
        v16bf ah, al;
#pragma unroll
        for (int j = 0; j < 8; ++j) {
            int kloc = kk * 32 + 2 * kbA + 2 * j;
            float a0 = zp[(size_t)(m0 + mrow) * HDIM + kloc];
            float a1 = zp[(size_t)(m0 + mrow) * HDIM + kloc + 1];
            __bf16 h0 = f2bf(a0), h1 = f2bf(a1);
            ah[2 * j] = h0;                  ah[2 * j + 1] = h1;
            al[2 * j] = f2bf(a0 - bf2f(h0)); al[2 * j + 1] = f2bf(a1 - bf2f(h1));
        }
#pragma unroll
        for (int ct = 0; ct < 8; ++ct) {
            int base = ((ct * 4 + kk) * 32 + lane) * 8;
            v16bf bh = __builtin_bit_cast(v16bf, *(const v8u*)(S1hi + base));
            v16bf bl = __builtin_bit_cast(v16bf, *(const v8u*)(S1lo + base));
            acc[ct] = __builtin_amdgcn_wmma_f32_16x16x32_bf16(false, ah, false, bh, (short)0, acc[ct], false, false);
            acc[ct] = __builtin_amdgcn_wmma_f32_16x16x32_bf16(false, ah, false, bl, (short)0, acc[ct], false, false);
            acc[ct] = __builtin_amdgcn_wmma_f32_16x16x32_bf16(false, al, false, bh, (short)0, acc[ct], false, false);
        }
    }

    // epilogue: tanh(+bias) * w2, reduce 128 cols per row, then reduce over rows into wsum[p]
    float rowacc[8] = {};
#pragma unroll
    for (int ct = 0; ct < 8; ++ct) {
        int col = ct * 16 + ncol;
        float bb = b1[col], ww = w2[col];
#pragma unroll
        for (int r = 0; r < 8; ++r)
            rowacc[r] += tanhf(acc[ct][r] + bb) * ww;
    }
    float tot = 0.f;
#pragma unroll
    for (int r = 0; r < 8; ++r) {
        float v = rowacc[r];
        v += __shfl_xor(v, 1, 32);
        v += __shfl_xor(v, 2, 32);
        v += __shfl_xor(v, 4, 32);
        v += __shfl_xor(v, 8, 32);   // per-16-lane-half row sums
        tot += v;
    }
    if (lane == 0 || lane == 16) atomicAdd(&wsum[p], tot);
}

// ---------------- beta = softmax(mean_n w) ----------------
__global__ void beta_kernel(const float* __restrict__ wsum, float* __restrict__ beta) {
    float m0 = wsum[0] / (float)NN, m1 = wsum[1] / (float)NN, m2 = wsum[2] / (float)NN;
    float mx = fmaxf(m0, fmaxf(m1, m2));
    float e0 = expf(m0 - mx), e1 = expf(m1 - mx), e2 = expf(m2 - mx);
    float s = e0 + e1 + e2;
    beta[0] = e0 / s; beta[1] = e1 / s; beta[2] = e2 / s;
}

// ---------------- out = sum_p beta[p] * z[p] ----------------
__global__ void output_kernel(const float* __restrict__ z, const float* __restrict__ beta,
                              float* __restrict__ out) {
    long long t = (long long)blockIdx.x * blockDim.x + threadIdx.x;
    if (t >= (long long)NN * HDIM) return;
    long long n = t / HDIM;
    int dim = (int)(t % HDIM);
    out[t] = beta[0] * z[n * HDIM + dim]
           + beta[1] * z[((size_t)NN + n) * HDIM + dim]
           + beta[2] * z[((size_t)2 * NN + n) * HDIM + dim];
}

extern "C" void kernel_launch(void* const* d_in, const int* in_sizes, int n_in,
                              void* d_out, int out_size, void* d_ws, size_t ws_size,
                              hipStream_t stream) {
    const float* h    = (const float*)d_in[0];
    const int*   src  = (const int*)d_in[1];
    const int*   dst  = (const int*)d_in[2];
    const float* W    = (const float*)d_in[3];
    const float* al   = (const float*)d_in[4];
    const float* ar   = (const float*)d_in[5];
    const float* bias = (const float*)d_in[6];
    const float* w1   = (const float*)d_in[7];
    const float* b1   = (const float*)d_in[8];
    const float* w2   = (const float*)d_in[9];
    float* out = (float*)d_out;

    const size_t PNHD = (size_t)PPATH * NN * HDIM;  // 38.4M
    const size_t PNH  = (size_t)PPATH * NN * HH;    // 1.2M
    const size_t PEH  = (size_t)PPATH * EDGES * HH; // 7.2M

    float* ws     = (float*)d_ws;
    float* feat   = ws;                  // PNHD
    float* z      = feat + PNHD;         // PNHD  (rst then elu'd in place)
    float* el     = z + PNHD;            // PNH
    float* er     = el + PNH;            // PNH
    unsigned* mk  = (unsigned*)(er + PNH);   // PNH
    float* denom  = (float*)(mk + PNH);  // PNH
    float* exbuf  = denom + PNH;         // PEH
    float* wsum   = exbuf + PEH;         // 8
    float* beta   = wsum + 8;            // 8
    unsigned* Whi = (unsigned*)(beta + 8);   // WPACK_N
    unsigned* Wlo = Whi + WPACK_N;           // WPACK_N
    unsigned* S1h = Wlo + WPACK_N;           // S1PACK_N
    unsigned* S1l = S1h + S1PACK_N;          // S1PACK_N

    // deterministic re-init every call
    fill_u32<<<2048, 256, 0, stream>>>((unsigned*)z, (long long)PNHD, 0u);
    fill_u32<<<256, 256, 0, stream>>>((unsigned*)denom, (long long)PNH, 0u);
    fill_u32<<<256, 256, 0, stream>>>(mk, (long long)PNH, 0x007FFFFFu);  // enc(-inf)
    fill_u32<<<1, 32, 0, stream>>>((unsigned*)wsum, 16, 0u);

    pack_W<<<(WPACK_N + 255) / 256, 256, 0, stream>>>(W, Whi, Wlo);
    pack_S1<<<(S1PACK_N + 255) / 256, 256, 0, stream>>>(w1, S1h, S1l);

    gemm_feat<<<dim3(NN / 16, PPATH), 256, 0, stream>>>(h, Whi, Wlo, feat);
    el_er_kernel<<<(int)((PNH + 255) / 256), 256, 0, stream>>>(feat, al, ar, el, er);
    edge_max_kernel<<<(PPATH * EDGES + 255) / 256, 256, 0, stream>>>(src, dst, el, er, mk);
    edge_exp_kernel<<<(PPATH * EDGES + 255) / 256, 256, 0, stream>>>(src, dst, el, er, mk, exbuf, denom);
    message_kernel<<<(int)(((long long)PPATH * EDGES * HDIM + 255) / 256), 256, 0, stream>>>(
        src, dst, feat, exbuf, denom, z);
    elu_kernel<<<(int)((PNHD + 255) / 256), 256, 0, stream>>>(z, bias);
    sa_kernel<<<(PPATH * (NN / 16) + 7) / 8, 256, 0, stream>>>(z, S1h, S1l, b1, w2, wsum);
    beta_kernel<<<1, 1, 0, stream>>>(wsum, beta);
    output_kernel<<<(int)(((long long)NN * HDIM + 255) / 256), 256, 0, stream>>>(z, beta, out);
}
